// Model_44787918962765
// MI455X (gfx1250) — matrix operand order repaired and checked by reading, not compiled, not run on hardware
//
#include <hip/hip_runtime.h>
#include <hip/hip_bf16.h>

typedef __attribute__((ext_vector_type(16))) _Float16 v16h;
typedef __attribute__((ext_vector_type(8)))  _Float16 v8h;
typedef __attribute__((ext_vector_type(8)))  float    v8f;

#define BN_SCALE 0.99999500003749986f  // 1/sqrt(1+1e-5)
#define KNN 10
#define NPTS 2048
#define NB 8

// ---------------------------------------------------------------------------
// WMMA tile loaders (CDNA5 16x16x32 f16 layouts, wave32)
// A (16xK row-major): lane L (m = L&15) holds K = kb..kb+7 and kb+16..kb+23,
// kb = 0 for lanes 0-15, 8 for lanes 16-31.  (ISA 05_wmma §7.12.2)
// B is consumed as Bt (N x K row-major, i.e. B transposed): lane L (n = L&15)
// holds K = kb..kb+7 and kb+16..kb+23, kb = 0 / 8 per half-wave (same as A).
// ---------------------------------------------------------------------------
__device__ __forceinline__ v16h load_a_tile(const _Float16* A0, int ldA, int kk, int lane) {
    const int m  = lane & 15;
    const int kb = (lane < 16) ? 0 : 8;
    const _Float16* p = A0 + (size_t)m * ldA + kk + kb;
    v8h lo = *reinterpret_cast<const v8h*>(p);
    v8h hi = *reinterpret_cast<const v8h*>(p + 16);
    v16h a;
#pragma unroll
    for (int i = 0; i < 8; ++i) { a[i] = lo[i]; a[i + 8] = hi[i]; }
    return a;
}

__device__ __forceinline__ v16h load_bt_tile(const _Float16* Bt0, int ldB, int kk, int lane) {
    const int n  = lane & 15;
    const int kb = (lane < 16) ? 0 : 8;
    const _Float16* p = Bt0 + (size_t)n * ldB + kk + kb;
    v8h lo = *reinterpret_cast<const v8h*>(p);
    v8h hi = *reinterpret_cast<const v8h*>(p + 16);
    v16h b;
#pragma unroll
    for (int i = 0; i < 8; ++i) { b[i] = lo[i]; b[i + 8] = hi[i]; }
    return b;
}

// ---------------------------------------------------------------------------
// Weight conversion: W (O x Kin f32) -> Wh (O x Kpad f16), zero pad K
// ---------------------------------------------------------------------------
__global__ void cvtw_kernel(const float* __restrict__ W, _Float16* __restrict__ Wh,
                            int Kin, int Kpad, int total) {
    int t = blockIdx.x * blockDim.x + threadIdx.x;
    if (t >= total) return;
    int k = t % Kpad, o = t / Kpad;
    Wh[t] = (k < Kin) ? (_Float16)W[(size_t)o * Kin + k] : (_Float16)0.f;
}

// x (B,N,3) f32 -> Xf16 slots (b*N*128) with row stride 32, zero padded
__global__ void buildx0_kernel(const float* __restrict__ x, _Float16* __restrict__ X, int total) {
    int t = blockIdx.x * blockDim.x + threadIdx.x;
    if (t >= total) return;
    int c  = t & 31;
    int bn = t >> 5;
    int b = bn / NPTS, n = bn % NPTS;
    _Float16 v = (c < 3) ? (_Float16)x[(size_t)bn * 3 + c] : (_Float16)0.f;
    X[(size_t)b * NPTS * 128 + (size_t)n * 32 + c] = v;
}

// per-point squared norms from f16 features (f32 accumulate)
__global__ void norms_kernel(const _Float16* __restrict__ X, float* __restrict__ xx, int Cs, int total) {
    int t = blockIdx.x * blockDim.x + threadIdx.x;
    if (t >= total) return;
    int b = t / NPTS, n = t % NPTS;
    const _Float16* row = X + (size_t)b * NPTS * 128 + (size_t)n * Cs;
    float s = 0.f;
    for (int c = 0; c < Cs; ++c) { float v = (float)row[c]; s += v * v; }
    xx[t] = s;
}

// ---------------------------------------------------------------------------
// D[i][j] = 2*<x_i,x_j> - ||x_i||^2 - ||x_j||^2 via WMMA (one wave per 16x16)
// ---------------------------------------------------------------------------
__global__ void __launch_bounds__(32) gram_dist_kernel(const _Float16* __restrict__ X,
                                                       const float* __restrict__ xx,
                                                       float* __restrict__ D, int Cs) {
    const int lane = threadIdx.x;
    const int mt = blockIdx.x, nt = blockIdx.y;
    const _Float16* A0 = X + (size_t)mt * 16 * Cs;
    const _Float16* B0 = X + (size_t)nt * 16 * Cs;
    v8f acc = {};
    for (int kk = 0; kk < Cs; kk += 32) {
        v16h a = load_a_tile(A0, Cs, kk, lane);
        v16h b = load_bt_tile(B0, Cs, kk, lane);
        acc = __builtin_amdgcn_wmma_f32_16x16x32_f16(false, a, false, b, (short)0, acc, false, false);
    }
    const int col   = nt * 16 + (lane & 15);
    const int rbase = mt * 16 + ((lane < 16) ? 0 : 8);
    const float xxc = xx[col];
#pragma unroll
    for (int r = 0; r < 8; ++r) {
        int row = rbase + r;
        D[(size_t)row * NPTS + col] = 2.0f * acc[r] - xx[row] - xxc;
    }
}

// top-10 (largest) per row, tie -> lowest index (strict >), like lax.top_k
__global__ void topk_kernel(const float* __restrict__ D, int* __restrict__ idx) {
    int row = blockIdx.x * blockDim.x + threadIdx.x;
    if (row >= NPTS) return;
    float bv[KNN]; int bi[KNN];
#pragma unroll
    for (int q = 0; q < KNN; ++q) { bv[q] = -3.4e38f; bi[q] = 0; }
    const float* dr = D + (size_t)row * NPTS;
    for (int j = 0; j < NPTS; ++j) {
        float v = dr[j];
        if (v > bv[KNN - 1]) {
            int p = KNN - 1;
            while (p > 0 && v > bv[p - 1]) { bv[p] = bv[p - 1]; bi[p] = bi[p - 1]; --p; }
            bv[p] = v; bi[p] = j;
        }
    }
#pragma unroll
    for (int q = 0; q < KNN; ++q) idx[row * KNN + q] = bi[q];
}

// Edge features: F[(n*10+j)][c] = concat(x_nbr - x_n, x_n), K padded to K2pad
__global__ void gather_kernel(const _Float16* __restrict__ X, const int* __restrict__ idx,
                              _Float16* __restrict__ F, int C, int Cs, int K2pad, int total) {
    int t = blockIdx.x * blockDim.x + threadIdx.x;
    if (t >= total) return;
    int c = t % K2pad;
    int r = t / K2pad;      // r = n*KNN + j
    int n = r / KNN;
    _Float16 v = (_Float16)0.f;
    if (c < C) {
        int nbr = idx[r];
        v = X[(size_t)nbr * Cs + c] - X[(size_t)n * Cs + c];
    } else if (c < 2 * C) {
        v = X[(size_t)n * Cs + (c - C)];
    }
    F[t] = v;
}

// Y (M x O, f32) = A (M x K f16) * Bt^T, Bt is (O x K f16) row-major weights
__global__ void __launch_bounds__(32) gemm_wmma_kernel(const _Float16* __restrict__ A,
                                                       const _Float16* __restrict__ Bt,
                                                       float* __restrict__ Y,
                                                       int K, int O) {
    const int lane = threadIdx.x;
    const int mt = blockIdx.x, nt = blockIdx.y;
    const _Float16* A0 = A  + (size_t)mt * 16 * K;
    const _Float16* B0 = Bt + (size_t)nt * 16 * K;
    v8f acc = {};
    for (int kk = 0; kk < K; kk += 32) {
        v16h a = load_a_tile(A0, K, kk, lane);
        v16h b = load_bt_tile(B0, K, kk, lane);
        acc = __builtin_amdgcn_wmma_f32_16x16x32_f16(false, a, false, b, (short)0, acc, false, false);
    }
    const int col   = nt * 16 + (lane & 15);
    const int rbase = mt * 16 + ((lane < 16) ? 0 : 8);
#pragma unroll
    for (int r = 0; r < 8; ++r) Y[(size_t)(rbase + r) * O + col] = acc[r];
}

// BN + LeakyReLU + max over k; scatter to d_out slice, next-layer f16, concat f16
__global__ void reduce_edge_kernel(const float* __restrict__ Y,
                                   const float* __restrict__ g, const float* __restrict__ bvec,
                                   float* __restrict__ outseg, _Float16* __restrict__ Xnext,
                                   int nextStride, _Float16* __restrict__ featrow, int O) {
    int t = blockIdx.x * blockDim.x + threadIdx.x;
    if (t >= NPTS * O) return;
    int o = t % O, n = t / O;
    float s = g[o] * BN_SCALE, bb = bvec[o];
    float m = -3.4e38f;
    const float* y = Y + (size_t)n * KNN * O + o;
#pragma unroll
    for (int j = 0; j < KNN; ++j) {
        float v = y[(size_t)j * O] * s + bb;
        v = (v >= 0.f) ? v : 0.2f * v;
        m = fmaxf(m, v);
    }
    outseg[(size_t)o * NPTS + n] = m;
    featrow[(size_t)n * 512 + o] = (_Float16)m;
    if (Xnext) Xnext[(size_t)n * nextStride + o] = (_Float16)m;
}

// y5 = W5 * xc, BN + LReLU, fused global max/avg over N (y5 never stored)
__global__ void __launch_bounds__(32) w5_glob_kernel(const _Float16* __restrict__ W5h,
                                                     const _Float16* __restrict__ feat,
                                                     const float* __restrict__ g5,
                                                     const float* __restrict__ b5,
                                                     float* __restrict__ globout) {
    const int lane = threadIdx.x;
    const int b = blockIdx.x, et = blockIdx.y;
    const int half = (lane < 16) ? 0 : 8;
    const _Float16* A0 = W5h + (size_t)et * 16 * 512;
    const _Float16* F0 = feat + (size_t)b * NPTS * 512;
    float s[8], bb[8], rmax[8], rsum[8];
#pragma unroll
    for (int r = 0; r < 8; ++r) {
        int e = et * 16 + half + r;
        s[r] = g5[e] * BN_SCALE; bb[r] = b5[e];
        rmax[r] = -3.4e38f; rsum[r] = 0.f;
    }
    for (int pt = 0; pt < NPTS / 16; ++pt) {
        const _Float16* B0 = F0 + (size_t)pt * 16 * 512;
        v8f acc = {};
        for (int kk = 0; kk < 512; kk += 32) {
            v16h a = load_a_tile(A0, 512, kk, lane);
            v16h bvv = load_bt_tile(B0, 512, kk, lane);
            acc = __builtin_amdgcn_wmma_f32_16x16x32_f16(false, a, false, bvv, (short)0, acc, false, false);
        }
#pragma unroll
        for (int r = 0; r < 8; ++r) {
            float y = acc[r] * s[r] + bb[r];
            y = (y >= 0.f) ? y : 0.2f * y;
            rmax[r] = fmaxf(rmax[r], y);
            rsum[r] += y;
        }
    }
    // reduce over the 16 point-columns (butterfly stays inside each half-wave)
#pragma unroll
    for (int off = 1; off < 16; off <<= 1) {
#pragma unroll
        for (int r = 0; r < 8; ++r) {
            rmax[r] = fmaxf(rmax[r], __shfl_xor(rmax[r], off, 32));
            rsum[r] += __shfl_xor(rsum[r], off, 32);
        }
    }
    if ((lane & 15) == 0) {
#pragma unroll
        for (int r = 0; r < 8; ++r) {
            int e = et * 16 + half + r;
            globout[(size_t)b * 2048 + e]        = rmax[r];
            globout[(size_t)b * 2048 + 1024 + e] = rsum[r] * (1.0f / NPTS);
        }
    }
}

// x_seg[:, 0:2048, :] = glob broadcast over N
__global__ void bcast_kernel(const float* __restrict__ glob, float* __restrict__ out, long total) {
    long t = (long)blockIdx.x * blockDim.x + threadIdx.x;
    if (t >= total) return;
    int n = (int)(t % NPTS);
    long be = t / NPTS;
    int e = (int)(be % 2048);
    int b = (int)(be / 2048);
    out[(size_t)b * 2560 * NPTS + (size_t)e * NPTS + n] = glob[(size_t)b * 2048 + e];
}

// ---------------------------------------------------------------------------
extern "C" void kernel_launch(void* const* d_in, const int* in_sizes, int n_in,
                              void* d_out, int out_size, void* d_ws, size_t ws_size,
                              hipStream_t stream) {
    (void)in_sizes; (void)n_in; (void)out_size; (void)ws_size;
    const float* x = (const float*)d_in[0];
    const float* Wsrc[5] = {(const float*)d_in[2], (const float*)d_in[5], (const float*)d_in[8],
                            (const float*)d_in[11], (const float*)d_in[14]};
    const float* Gv[5]   = {(const float*)d_in[3], (const float*)d_in[6], (const float*)d_in[9],
                            (const float*)d_in[12], (const float*)d_in[15]};
    const float* Bv[5]   = {(const float*)d_in[4], (const float*)d_in[7], (const float*)d_in[10],
                            (const float*)d_in[13], (const float*)d_in[16]};

    char* ws = (char*)d_ws;
    size_t off = 0;
    auto alloc = [&](size_t bytes) -> void* {
        void* p = ws + off;
        off = (off + bytes + 255) & ~(size_t)255;
        return p;
    };
    const int WKin[5]  = {6, 128, 128, 256, 512};
    const int WKpad[5] = {32, 128, 128, 256, 512};
    const int WO[5]    = {64, 64, 128, 256, 1024};
    _Float16* Wh[5];
    for (int i = 0; i < 5; ++i) Wh[i] = (_Float16*)alloc((size_t)WO[i] * WKpad[i] * 2);
    _Float16* Xf   = (_Float16*)alloc((size_t)NB * NPTS * 128 * 2);
    _Float16* feat = (_Float16*)alloc((size_t)NB * NPTS * 512 * 2);
    float*    xx   = (float*)alloc((size_t)NB * NPTS * 4);
    float*    Dm   = (float*)alloc((size_t)NPTS * NPTS * 4);
    int*      idx  = (int*)alloc((size_t)NPTS * KNN * 4);
    _Float16* Ff   = (_Float16*)alloc((size_t)NPTS * KNN * 256 * 2);
    float*    Yg   = (float*)alloc((size_t)NPTS * KNN * 256 * 4);

    // f16 weights + layer-0 features
    for (int i = 0; i < 5; ++i) {
        int total = WO[i] * WKpad[i];
        cvtw_kernel<<<(total + 255) / 256, 256, 0, stream>>>(Wsrc[i], Wh[i], WKin[i], WKpad[i], total);
    }
    {
        int total = NB * NPTS * 32;
        buildx0_kernel<<<(total + 255) / 256, 256, 0, stream>>>(x, Xf, total);
    }

    struct Cfg { int C, Cs, K2, O, nextCs, segoff; };
    const Cfg L[4] = {
        {  3,  32,  32,  64,  64,   0},
        { 64,  64, 128,  64,  64,  64},
        { 64,  64, 128, 128, 128, 128},
        {128, 128, 256, 256,   0, 256},
    };

    for (int l = 0; l < 4; ++l) {
        norms_kernel<<<(NB * NPTS + 255) / 256, 256, 0, stream>>>(Xf, xx, L[l].Cs, NB * NPTS);
        for (int b = 0; b < NB; ++b) {
            const _Float16* Xb = Xf + (size_t)b * NPTS * 128;
            const float* xxb = xx + (size_t)b * NPTS;
            gram_dist_kernel<<<dim3(NPTS / 16, NPTS / 16), 32, 0, stream>>>(Xb, xxb, Dm, L[l].Cs);
            topk_kernel<<<NPTS / 256, 256, 0, stream>>>(Dm, idx);
            int gtotal = NPTS * KNN * L[l].K2;
            gather_kernel<<<(gtotal + 255) / 256, 256, 0, stream>>>(Xb, idx, Ff, L[l].C, L[l].Cs, L[l].K2, gtotal);
            gemm_wmma_kernel<<<dim3(NPTS * KNN / 16, L[l].O / 16), 32, 0, stream>>>(Ff, Wh[l], Yg, L[l].K2, L[l].O);
            float* outseg = (float*)d_out + (size_t)b * 2560 * NPTS + (size_t)(2048 + L[l].segoff) * NPTS;
            _Float16* Xn = (l < 3) ? (Xf + (size_t)b * NPTS * 128) : nullptr;
            _Float16* featrow = feat + (size_t)b * NPTS * 512 + L[l].segoff;
            reduce_edge_kernel<<<(NPTS * L[l].O + 255) / 256, 256, 0, stream>>>(
                Yg, Gv[l], Bv[l], outseg, Xn, L[l].nextCs, featrow, L[l].O);
        }
    }

    float* glob = (float*)d_out + (size_t)NB * 2560 * NPTS;
    w5_glob_kernel<<<dim3(NB, 64), 32, 0, stream>>>(Wh[4], feat, Gv[4], Bv[4], glob);
    long btotal = (long)NB * 2048 * NPTS;
    bcast_kernel<<<(unsigned)((btotal + 255) / 256), 256, 0, stream>>>(glob, (float*)d_out, btotal);
}
